// QuantumGraphNeuralNetwork_65481071403153
// MI455X (gfx1250) — matrix operand order repaired and checked by reading, not compiled, not run on hardware
//
#include <hip/hip_runtime.h>
#include <hip/hip_bf16.h>

// ---------------------------------------------------------------------------
// QuantumGraphNeuralNetwork: complex Gram matrix |<i|j>|^2 + thresholded graph
//
// Strategy (MI455X / gfx1250):
//  * Split-bf16 ("3xBF16") emulated fp32 GEMM on v_wmma_f32_16x16x32_bf16:
//    x = hi(bf16) + lo(bf16);  x*y ~= hi*hi + hi*lo + lo*hi  (~fp32 accuracy,
//    required since fid feeds 0.8 / 0.5 threshold comparisons). bf16 WMMA runs
//    ~8x the fp32 16x16x4 WMMA rate, so 3 GEMMs at bf16 beat 1 GEMM at fp32.
//  * fid is symmetric (g_re sym, g_im antisym) -> compute only upper-tri
//    128x128 tiles, mirror fid, zero-fill lower-tri weights (halves matmuls).
//  * g_im = im@re^T - re@im^T accumulated as (pos - neg) pair since bf16 WMMA
//    NEG modifiers only support C-negate per ISA 7.12.
//  * Software pipeline: prefetch next K-chunk's fp32 data into VGPRs while the
//    current chunk's 96 WMMAs execute; s_wait_loadcnt lands at the next
//    iteration's convert, hiding HBM latency behind the matrix pipe.
// ---------------------------------------------------------------------------

typedef __attribute__((ext_vector_type(16))) __bf16 v16bf;
typedef __attribute__((ext_vector_type(8)))  float  v8f;

#define WMMA_BF16(A, B, C) \
    __builtin_amdgcn_wmma_f32_16x16x32_bf16(false, (A), false, (B), (short)0, (C), false, false)

static constexpr int N  = 8192;   // samples
static constexpr int D  = 4096;   // statevector dim
static constexpr int BT = 128;    // output tile (BT x BT)
static constexpr int KT = 32;     // K chunk per WMMA step
static constexpr int SEG = BT * KT;  // one LDS panel: 128 rows x 32 halves

// Split 16 fp32 values (held in registers) into truncated-bf16 hi + bf16 lo
// correction, store both as 2x b128 into LDS panels.
__device__ __forceinline__ void split16reg(const float4 f4[4],
                                           unsigned short* __restrict__ dhi,
                                           unsigned short* __restrict__ dlo) {
    alignas(16) unsigned short hi[16];
    alignas(16) unsigned short lo[16];
#pragma unroll
    for (int q = 0; q < 4; ++q) {
        float vals[4] = {f4[q].x, f4[q].y, f4[q].z, f4[q].w};
#pragma unroll
        for (int e = 0; e < 4; ++e) {
            unsigned int xb = __float_as_uint(vals[e]);
            unsigned short h = (unsigned short)(xb >> 16);          // truncate
            float hf = __uint_as_float(((unsigned int)h) << 16);
            float lf = vals[e] - hf;                                // exact
            unsigned short l = (unsigned short)(__float_as_uint(lf) >> 16);
            hi[q * 4 + e] = h;
            lo[q * 4 + e] = l;
        }
    }
    ((uint4*)dhi)[0] = ((const uint4*)hi)[0];
    ((uint4*)dhi)[1] = ((const uint4*)hi)[1];
    ((uint4*)dlo)[0] = ((const uint4*)lo)[0];
    ((uint4*)dlo)[1] = ((const uint4*)lo)[1];
}

#define LOADP(dst, src, k0)                                        \
    do {                                                           \
        const float4* _s = (const float4*)((src) + (k0));          \
        (dst)[0] = _s[0]; (dst)[1] = _s[1];                        \
        (dst)[2] = _s[2]; (dst)[3] = _s[3];                        \
    } while (0)

// A fragment (16x32 bf16, ISA 7.12.2): lane L (row = L%16):
//   VGPR0-3 = K {0..7} (L<16) or {8..15} (L>=16), VGPR4-7 = +16.
__device__ __forceinline__ v16bf loadA(const unsigned short* __restrict__ base,
                                       int row16, int lane) {
    int off = (row16 + (lane & 15)) * KT + ((lane & 16) ? 8 : 0);
    union { uint4 q[2]; v16bf v; } u;
    u.q[0] = *(const uint4*)(base + off);
    u.q[1] = *(const uint4*)(base + off + 16);
    return u.v;
}

// B fragment (32x16 bf16): lane n (col = n%16) holds K = (n/16)*8 + [0..7], +16,
// packed 2-per-dword. Panel is stored row-major per COLUMN (rows of latent),
// so each lane reads two runs of 8 contiguous halves, 16 halves apart.
__device__ __forceinline__ v16bf loadB(const unsigned short* __restrict__ base,
                                       int col16, int lane) {
    int off = (col16 + (lane & 15)) * KT + ((lane & 16) ? 8 : 0);
    union { uint4 q[2]; v16bf v; } u;
    u.q[0] = *(const uint4*)(base + off);
    u.q[1] = *(const uint4*)(base + off + 16);
    return u.v;
}

__global__ __launch_bounds__(256)
void qgnn_gram_tile_kernel(const float* __restrict__ lre,
                           const float* __restrict__ lim,
                           float* __restrict__ w_out,
                           float* __restrict__ fid_out) {
    const int ti = blockIdx.y;   // row tile
    const int tj = blockIdx.x;   // col tile
    if (tj < ti) return;         // symmetric: upper-triangular tiles only

    // 64 KB LDS: 8 panels of 128x32 bf16 (A/B sides, re/im, hi/lo)
    __shared__ unsigned short lds[8 * SEG];
    unsigned short* AreH = lds + 0 * SEG;
    unsigned short* AreL = lds + 1 * SEG;
    unsigned short* AimH = lds + 2 * SEG;
    unsigned short* AimL = lds + 3 * SEG;
    unsigned short* BreH = lds + 4 * SEG;
    unsigned short* BreL = lds + 5 * SEG;
    unsigned short* BimH = lds + 6 * SEG;
    unsigned short* BimL = lds + 7 * SEG;

    const int t    = threadIdx.x;
    const int lane = t & 31;
    const int wave = t >> 5;        // 8 waves
    const int wr   = wave >> 1;     // 0..3 -> 32-row band
    const int wc   = wave & 1;      // 0..1 -> 64-col band

    const int riBase = ti * BT;
    const int rjBase = tj * BT;

    // Accumulators: g_re and g_im = (pos - neg), 2x4 fragments per wave.
    v8f z = {0.f, 0.f, 0.f, 0.f, 0.f, 0.f, 0.f, 0.f};
    v8f gre[2][4], gp[2][4], gn[2][4];
#pragma unroll
    for (int r = 0; r < 2; ++r)
#pragma unroll
        for (int c = 0; c < 4; ++c) { gre[r][c] = z; gp[r][c] = z; gn[r][c] = z; }

    // Cooperative loader role: thread t owns panel row r, K-half `half`.
    const int r    = t >> 1;   // 0..127
    const int half = t & 1;    // 16-element half of the 32-wide K chunk
    const int dst  = r * KT + half * 16;

    const float* srcAre = lre + (size_t)(riBase + r) * D + half * 16;
    const float* srcAim = lim + (size_t)(riBase + r) * D + half * 16;
    const float* srcBre = lre + (size_t)(rjBase + r) * D + half * 16;
    const float* srcBim = lim + (size_t)(rjBase + r) * D + half * 16;

    // Prefetch registers for the software pipeline (16 float4 = 64 VGPRs).
    float4 pAre[4], pAim[4], pBre[4], pBim[4];
    LOADP(pAre, srcAre, 0);
    LOADP(pAim, srcAim, 0);
    LOADP(pBre, srcBre, 0);
    LOADP(pBim, srcBim, 0);

    const int nch = D / KT;
    for (int kk = 0; kk < nch; ++kk) {
        // Convert previously-fetched chunk into bf16 hi/lo LDS panels.
        split16reg(pAre, AreH + dst, AreL + dst);
        split16reg(pAim, AimH + dst, AimL + dst);
        split16reg(pBre, BreH + dst, BreL + dst);
        split16reg(pBim, BimH + dst, BimL + dst);
        __syncthreads();

        // Issue next chunk's global loads; they complete during the WMMAs.
        if (kk + 1 < nch) {
            const int k0n = (kk + 1) * KT;
            LOADP(pAre, srcAre, k0n);
            LOADP(pAim, srcAim, k0n);
            LOADP(pBre, srcBre, k0n);
            LOADP(pBim, srcBim, k0n);
        }

        v16bf arH[2], arL[2], aiH[2], aiL[2];
#pragma unroll
        for (int rr = 0; rr < 2; ++rr) {
            const int row16 = wr * 32 + rr * 16;
            arH[rr] = loadA(AreH, row16, lane);
            arL[rr] = loadA(AreL, row16, lane);
            aiH[rr] = loadA(AimH, row16, lane);
            aiL[rr] = loadA(AimL, row16, lane);
        }

#pragma unroll
        for (int c = 0; c < 4; ++c) {
            const int col16 = wc * 64 + c * 16;
            v16bf brH = loadB(BreH, col16, lane);
            v16bf brL = loadB(BreL, col16, lane);
            v16bf biH = loadB(BimH, col16, lane);
            v16bf biL = loadB(BimL, col16, lane);
#pragma unroll
            for (int rr = 0; rr < 2; ++rr) {
                // g_re += re_i . re_j  (3xBF16)
                gre[rr][c] = WMMA_BF16(arH[rr], brH, gre[rr][c]);
                gre[rr][c] = WMMA_BF16(arH[rr], brL, gre[rr][c]);
                gre[rr][c] = WMMA_BF16(arL[rr], brH, gre[rr][c]);
                // g_re += im_i . im_j
                gre[rr][c] = WMMA_BF16(aiH[rr], biH, gre[rr][c]);
                gre[rr][c] = WMMA_BF16(aiH[rr], biL, gre[rr][c]);
                gre[rr][c] = WMMA_BF16(aiL[rr], biH, gre[rr][c]);
                // g_im(+) += im_i . re_j
                gp[rr][c] = WMMA_BF16(aiH[rr], brH, gp[rr][c]);
                gp[rr][c] = WMMA_BF16(aiH[rr], brL, gp[rr][c]);
                gp[rr][c] = WMMA_BF16(aiL[rr], brH, gp[rr][c]);
                // g_im(-) += re_i . im_j
                gn[rr][c] = WMMA_BF16(arH[rr], biH, gn[rr][c]);
                gn[rr][c] = WMMA_BF16(arH[rr], biL, gn[rr][c]);
                gn[rr][c] = WMMA_BF16(arL[rr], biH, gn[rr][c]);
            }
        }
        __syncthreads();   // all LDS readers done before next convert
    }

    // Writeback. C/D layout: VGPR v, lanes 0-15 -> M=v, lanes 16-31 -> M=v+8.
    const int nLoc = lane & 15;
    const int mOff = (lane >> 4) * 8;
#pragma unroll
    for (int rr = 0; rr < 2; ++rr) {
#pragma unroll
        for (int c = 0; c < 4; ++c) {
            v8f gim = gp[rr][c] - gn[rr][c];
#pragma unroll
            for (int v = 0; v < 8; ++v) {
                float grev = gre[rr][c][v];
                float gimv = gim[v];
                float fidv = grev * grev + gimv * gimv;
                const int gi = riBase + wr * 32 + rr * 16 + mOff + v;
                const int gj = rjBase + wc * 64 + c * 16 + nLoc;
                const size_t idx = (size_t)gi * N + gj;
                fid_out[idx] = fidv;
                float wv = 0.0f;
                if (gi < gj)
                    wv = (fidv >= 0.8f) ? 1.0f : ((fidv >= 0.5f) ? 0.5f : 0.0f);
                w_out[idx] = wv;
                if (ti != tj) {   // mirror: fid symmetric, lower weights = 0
                    const size_t idxT = (size_t)gj * N + gi;
                    fid_out[idxT] = fidv;
                    w_out[idxT]   = 0.0f;
                }
            }
        }
    }
}

extern "C" void kernel_launch(void* const* d_in, const int* in_sizes, int n_in,
                              void* d_out, int out_size, void* d_ws, size_t ws_size,
                              hipStream_t stream) {
    (void)in_sizes; (void)n_in; (void)out_size; (void)d_ws; (void)ws_size;
    const float* lre = (const float*)d_in[0];
    const float* lim = (const float*)d_in[1];
    float* out   = (float*)d_out;
    float* w     = out;                       // weights [N, N]
    float* fid   = out + (size_t)N * N;       // fid     [N, N]

    dim3 grid(N / BT, N / BT);                // 64 x 64 tiles, upper-tri active
    qgnn_gram_tile_kernel<<<grid, 256, 0, stream>>>(lre, lim, w, fid);
}
